// USTKT_82325933130307
// MI455X (gfx1250) — compile-verified, matrix operand order checked by reading, not run on hardware
//
#include <hip/hip_runtime.h>

// ---------------------------------------------------------------------------
// Problem constants (from reference): B=8, H=8, S=1024, DK=64, D=512
// ---------------------------------------------------------------------------
#define Bb  8
#define Hh  8
#define Sq  1024
#define DKc 64
#define Dd  512

typedef __attribute__((ext_vector_type(16))) __bf16 v16bf;
typedef __attribute__((ext_vector_type(8)))  float  v8f;

// ---------------------------------------------------------------------------
// Helpers
// ---------------------------------------------------------------------------
__device__ __forceinline__ v8f wmma_bf16(v16bf a, v16bf b, v8f c) {
  return __builtin_amdgcn_wmma_f32_16x16x32_bf16(false, a, false, b,
                                                 (short)0, c, false, false);
}

// Async global->LDS copy of one 16-byte chunk (CDNA5 ASYNCcnt path).
// LDS operand is the 32-bit wave-relative offset (low 32 bits of the generic
// address of a __shared__ pointer).
__device__ __forceinline__ void async_b128(const __bf16* lds, const __bf16* g) {
  unsigned l = (unsigned)(size_t)(const void*)lds;
  asm volatile("global_load_async_to_lds_b128 %0, %1, off"
               :: "v"(l), "v"(g) : "memory");
}
__device__ __forceinline__ void wait_async0() {
  asm volatile("s_wait_asynccnt 0" ::: "memory");
}

// Load one 16x32 bf16 fragment from a row-major LDS tile (pitch 40 bf16).
// lanes 0-15: rows 0-15, K in {0..7,16..23}; lanes 16-31: K in {8..15,24..31}.
__device__ __forceinline__ void load_frag(v16bf& f, const __bf16* rowp, int hb) {
  const __bf16* p0 = rowp + 8 * hb;
  const __bf16* p1 = rowp + 16 + 8 * hb;
#pragma unroll
  for (int e = 0; e < 8; ++e) {
    f[e]     = p0[e];
    f[8 + e] = p1[e];
  }
}

__device__ __forceinline__ void bsplit(float x, __bf16* hi, __bf16* lo) {
  __bf16 h = (__bf16)x;
  *hi = h;
  *lo = (__bf16)(x - (float)h);
}

// ---------------------------------------------------------------------------
// Kernel 0a: split fp32 -> bf16 hi/lo planes (elementwise, x4 vectorized)
// ---------------------------------------------------------------------------
__global__ __launch_bounds__(256) void split_kernel(
    const float* __restrict__ x, __bf16* __restrict__ hi,
    __bf16* __restrict__ lo, int n4) {
  int i = blockIdx.x * 256 + threadIdx.x;
  if (i >= n4) return;
  const float4 v = ((const float4*)x)[i];
  bsplit(v.x, hi + 4 * i + 0, lo + 4 * i + 0);
  bsplit(v.y, hi + 4 * i + 1, lo + 4 * i + 1);
  bsplit(v.z, hi + 4 * i + 2, lo + 4 * i + 2);
  bsplit(v.w, hi + 4 * i + 3, lo + 4 * i + 3);
}

// ---------------------------------------------------------------------------
// Kernel 0b: transpose + split weights: Wt[n][k] = W[k][n], bf16 hi/lo planes
// ---------------------------------------------------------------------------
__global__ __launch_bounds__(256) void wtrans_kernel(
    const float* __restrict__ W, __bf16* __restrict__ Wth,
    __bf16* __restrict__ Wtl) {
  __shared__ float t[32][33];
  const int k0 = blockIdx.x * 32, n0 = blockIdx.y * 32;
  const int tid = threadIdx.x;
  const int rr = tid >> 5, cc = tid & 31;
#pragma unroll
  for (int it = 0; it < 4; ++it)
    t[rr + it * 8][cc] = W[(size_t)(k0 + rr + it * 8) * Dd + n0 + cc];
  __syncthreads();
#pragma unroll
  for (int it = 0; it < 4; ++it) {
    const float v = t[cc][rr + it * 8];
    const size_t o = (size_t)(n0 + rr + it * 8) * Dd + k0 + cc;
    bsplit(v, Wth + o, Wtl + o);
  }
}

// ---------------------------------------------------------------------------
// Kernel 1: input projection  Y = X @ W + b  (bf16x3 split GEMM).
// A/B staged with global_load_async_to_lds_b128 from pre-split planes.
// Output fp32, head-major [B,H,S,DK].
// ---------------------------------------------------------------------------
__global__ __launch_bounds__(256) void proj_kernel(
    const __bf16* __restrict__ Xh, const __bf16* __restrict__ Xl,
    const __bf16* __restrict__ Wth, const __bf16* __restrict__ Wtl,
    const float* __restrict__ bias, float* __restrict__ Y) {
  const int m0 = blockIdx.x * 128;
  const int n0 = blockIdx.y * 64;
  __shared__ __bf16 Ah[128 * 40], Al[128 * 40];
  __shared__ __bf16 Bh[64 * 40],  Bl[64 * 40];
  const int tid = threadIdx.x;
  const int lane = tid & 31, wave = tid >> 5;
  const int wm = (wave & 3) * 32, wn = (wave >> 2) * 32;
  const int lr = lane & 15, hb = lane >> 4;
  const v8f vz = {0.f, 0.f, 0.f, 0.f, 0.f, 0.f, 0.f, 0.f};
  v8f acc[2][2] = {{vz, vz}, {vz, vz}};

  for (int kt = 0; kt < Dd / 32; ++kt) {
    // A tile: 128 rows x 32 bf16 = 512 chunks of 16B per plane
#pragma unroll
    for (int i = 0; i < 2; ++i) {
      const int c = tid + i * 256;
      const int r = c >> 2, q8 = (c & 3) << 3;
      const size_t g = (size_t)(m0 + r) * Dd + kt * 32 + q8;
      async_b128(Ah + r * 40 + q8, Xh + g);
      async_b128(Al + r * 40 + q8, Xl + g);
    }
    // B tile: 64 rows x 32 bf16 = 256 chunks per plane
    {
      const int n = tid >> 2, q8 = (tid & 3) << 3;
      const size_t g = (size_t)(n0 + n) * Dd + kt * 32 + q8;
      async_b128(Bh + n * 40 + q8, Wth + g);
      async_b128(Bl + n * 40 + q8, Wtl + g);
    }
    wait_async0();
    __syncthreads();

    v16bf ah[2], al2[2], bh2[2], bl2[2];
#pragma unroll
    for (int f = 0; f < 2; ++f) {
      load_frag(ah[f],  Ah + (wm + f * 16 + lr) * 40, hb);
      load_frag(al2[f], Al + (wm + f * 16 + lr) * 40, hb);
      load_frag(bh2[f], Bh + (wn + f * 16 + lr) * 40, hb);
      load_frag(bl2[f], Bl + (wn + f * 16 + lr) * 40, hb);
    }
#pragma unroll
    for (int fm = 0; fm < 2; ++fm)
#pragma unroll
      for (int fn = 0; fn < 2; ++fn) {
        acc[fm][fn] = wmma_bf16(ah[fm],  bh2[fn], acc[fm][fn]);
        acc[fm][fn] = wmma_bf16(ah[fm],  bl2[fn], acc[fm][fn]);
        acc[fm][fn] = wmma_bf16(al2[fm], bh2[fn], acc[fm][fn]);
      }
    __syncthreads();
  }
#pragma unroll
  for (int fm = 0; fm < 2; ++fm)
#pragma unroll
    for (int fn = 0; fn < 2; ++fn)
#pragma unroll
      for (int v = 0; v < 8; ++v) {
        int m = m0 + wm + fm * 16 + hb * 8 + v;
        int n = n0 + wn + fn * 16 + lr;
        float val = acc[fm][fn][v] + bias[n];
        int b = m >> 10, s = m & 1023;
        int h = n >> 6,  dk = n & 63;
        Y[((((size_t)b * Hh + h) << 10) + (size_t)s) * 64 + dk] = val;
      }
}

// ---------------------------------------------------------------------------
// Kernel 2: per-row stats + bf16 planes for the attention GEMMs.
// rowQ = sum(qm^2)+sum(qc_raw); rowK = sum(km^2)+sum(kc_raw)
// qm/sqrt(clip(qc))/km/sqrt(clip(kc)) -> head-major hi/lo planes;
// vm/vc -> per-head transposed planes Vt[bh][d][s].
// ---------------------------------------------------------------------------
__global__ __launch_bounds__(64) void prep_kernel(
    const float* __restrict__ qm, const float* __restrict__ qc,
    const float* __restrict__ km, const float* __restrict__ kc,
    const float* __restrict__ vm, const float* __restrict__ vc,
    __bf16* __restrict__ qmh, __bf16* __restrict__ qml,
    __bf16* __restrict__ sqh, __bf16* __restrict__ sql,
    __bf16* __restrict__ kmh, __bf16* __restrict__ kml,
    __bf16* __restrict__ skh, __bf16* __restrict__ skl,
    __bf16* __restrict__ vmth, __bf16* __restrict__ vmtl,
    __bf16* __restrict__ vcth, __bf16* __restrict__ vctl,
    float* __restrict__ rowQ, float* __restrict__ rowK) {
  __shared__ float r1[64], r2[64];
  const int r = blockIdx.x;
  const int d = threadIdx.x;
  const size_t o = (size_t)r * 64 + d;
  const float qmv = qm[o], qcv = qc[o], kmv = km[o], kcv = kc[o];
  const float vmv = vm[o], vcv = vc[o];
  r1[d] = qmv * qmv + qcv;
  r2[d] = kmv * kmv + kcv;
  bsplit(qmv, qmh + o, qml + o);
  bsplit(sqrtf(fmaxf(qcv, 1e-24f)), sqh + o, sql + o);
  bsplit(kmv, kmh + o, kml + o);
  bsplit(sqrtf(fmaxf(kcv, 1e-24f)), skh + o, skl + o);
  const int bh = r >> 10, s = r & 1023;
  const size_t to = ((size_t)bh * 64 + d) * 1024 + s;
  bsplit(vmv, vmth + to, vmtl + to);
  bsplit(vcv, vcth + to, vctl + to);
  __syncthreads();
  for (int off = 32; off > 0; off >>= 1) {
    if (d < off) { r1[d] += r1[d + off]; r2[d] += r2[d + off]; }
    __syncthreads();
  }
  if (d == 0) { rowQ[r] = r1[0]; rowK[r] = r2[0]; }
}

// ---------------------------------------------------------------------------
// Kernel 3: scores = (2*([qm|sq].[km|sk]^T) - rowQ - rowK)/8 per head.
// NT GEMM, K=128, causal block skip, async-staged bf16 planes.
// ---------------------------------------------------------------------------
__global__ __launch_bounds__(256) void scores_kernel(
    const __bf16* __restrict__ qmh, const __bf16* __restrict__ qml,
    const __bf16* __restrict__ sqh, const __bf16* __restrict__ sql,
    const __bf16* __restrict__ kmh, const __bf16* __restrict__ kml,
    const __bf16* __restrict__ skh, const __bf16* __restrict__ skl,
    const float* __restrict__ rowQ, const float* __restrict__ rowK,
    float* __restrict__ sc) {
  const int q0 = blockIdx.x * 128;
  const int k0 = blockIdx.y * 64;
  const int bh = blockIdx.z;
  if (k0 > q0 + 127) return;  // strictly above causal diagonal
  __shared__ __bf16 Ah[128 * 40], Al[128 * 40];
  __shared__ __bf16 Bh[64 * 40],  Bl[64 * 40];
  const int tid = threadIdx.x;
  const int lane = tid & 31, wave = tid >> 5;
  const int wm = (wave & 3) * 32, wn = (wave >> 2) * 32;
  const int lr = lane & 15, hb = lane >> 4;
  const v8f vz = {0.f, 0.f, 0.f, 0.f, 0.f, 0.f, 0.f, 0.f};
  v8f acc[2][2] = {{vz, vz}, {vz, vz}};

  const size_t hbase = (size_t)bh * Sq * 64;
  for (int kt = 0; kt < 4; ++kt) {
    const __bf16* Axh = (kt < 2) ? qmh : sqh;
    const __bf16* Axl = (kt < 2) ? qml : sql;
    const __bf16* Bxh = (kt < 2) ? kmh : skh;
    const __bf16* Bxl = (kt < 2) ? kml : skl;
    const int dbase = (kt & 1) * 32;
#pragma unroll
    for (int i = 0; i < 2; ++i) {
      const int c = tid + i * 256;
      const int r = c >> 2, q8 = (c & 3) << 3;
      const size_t g = hbase + (size_t)(q0 + r) * 64 + dbase + q8;
      async_b128(Ah + r * 40 + q8, Axh + g);
      async_b128(Al + r * 40 + q8, Axl + g);
    }
    {
      const int n = tid >> 2, q8 = (tid & 3) << 3;
      const size_t g = hbase + (size_t)(k0 + n) * 64 + dbase + q8;
      async_b128(Bh + n * 40 + q8, Bxh + g);
      async_b128(Bl + n * 40 + q8, Bxl + g);
    }
    wait_async0();
    __syncthreads();

    v16bf ah[2], al2[2], bh2[2], bl2[2];
#pragma unroll
    for (int f = 0; f < 2; ++f) {
      load_frag(ah[f],  Ah + (wm + f * 16 + lr) * 40, hb);
      load_frag(al2[f], Al + (wm + f * 16 + lr) * 40, hb);
      load_frag(bh2[f], Bh + (wn + f * 16 + lr) * 40, hb);
      load_frag(bl2[f], Bl + (wn + f * 16 + lr) * 40, hb);
    }
#pragma unroll
    for (int fm = 0; fm < 2; ++fm)
#pragma unroll
      for (int fn = 0; fn < 2; ++fn) {
        acc[fm][fn] = wmma_bf16(ah[fm],  bh2[fn], acc[fm][fn]);
        acc[fm][fn] = wmma_bf16(ah[fm],  bl2[fn], acc[fm][fn]);
        acc[fm][fn] = wmma_bf16(al2[fm], bh2[fn], acc[fm][fn]);
      }
    __syncthreads();
  }
#pragma unroll
  for (int fm = 0; fm < 2; ++fm)
#pragma unroll
    for (int fn = 0; fn < 2; ++fn)
#pragma unroll
      for (int v = 0; v < 8; ++v) {
        int q = q0 + wm + fm * 16 + hb * 8 + v;
        int k = k0 + wn + fn * 16 + lr;
        float val =
            (2.f * acc[fm][fn][v] - rowQ[bh * Sq + q] - rowK[bh * Sq + k]) * 0.125f;
        sc[(((size_t)bh << 10) + q) * 1024 + k] = val;
      }
}

// ---------------------------------------------------------------------------
// Kernel 4: per-row softmax -> cumsum -> position decay -> second softmax.
// One 256-thread block per (bh,q) row; written back in place
// (zeros for k>q; full zero row for q==0).
// ---------------------------------------------------------------------------
__global__ __launch_bounds__(256) void decay_softmax_kernel(
    float* __restrict__ sc, const float* __restrict__ gammas) {
  const int q = blockIdx.x, bh = blockIdx.y, h = bh & (Hh - 1);
  float* row = sc + (((size_t)bh << 10) + q) * 1024;
  __shared__ float srow[1024];
  __shared__ float trow[1024];
  __shared__ float red[256];
  __shared__ float sh_carry;
  const int tid = threadIdx.x;
  const int L = q + 1;

  float g = gammas[h];
  float sp = (g > 0.f) ? (g + log1pf(expf(-g))) : log1pf(expf(g));  // softplus
  const float gamma = -sp;

  for (int k = tid; k < 1024; k += 256) srow[k] = row[k];
  __syncthreads();

  float m = -3.0e38f;
  for (int k = tid; k < L; k += 256) m = fmaxf(m, srow[k]);
  red[tid] = m;
  __syncthreads();
  for (int off = 128; off > 0; off >>= 1) {
    if (tid < off) red[tid] = fmaxf(red[tid], red[tid + off]);
    __syncthreads();
  }
  const float m1 = red[0];
  __syncthreads();
  float s = 0.f;
  for (int k = tid; k < L; k += 256) s += expf(srow[k] - m1);
  red[tid] = s;
  __syncthreads();
  for (int off = 128; off > 0; off >>= 1) {
    if (tid < off) red[tid] += red[tid + off];
    __syncthreads();
  }
  const float inv1 = 1.f / red[0];
  __syncthreads();

  if (tid == 0) sh_carry = 0.f;
  __syncthreads();

  for (int base = 0; base < 1024; base += 256) {
    const int k = base + tid;
    const float p1 = (k < L) ? expf(srow[k] - m1) * inv1 : 0.f;
    const float cold = sh_carry;
    red[tid] = p1;
    __syncthreads();
    for (int off = 1; off < 256; off <<= 1) {
      float v = (tid >= off) ? red[tid - off] : 0.f;
      __syncthreads();
      red[tid] += v;
      __syncthreads();
    }
    const float incl = red[tid] + cold;
    const float pos  = fabsf((float)(q - k));
    const float dsv  = sqrtf(fmaxf((1.f - incl) * pos, 0.f));
    float te = expf(dsv * gamma);
    te = fminf(fmaxf(te, 1e-5f), 1e5f);
    trow[k] = (k < L) ? srow[k] * te : 0.f;
    __syncthreads();
    if (tid == 0) sh_carry = cold + red[255];
    __syncthreads();
  }

  m = -3.0e38f;
  for (int k = tid; k < L; k += 256) m = fmaxf(m, trow[k]);
  red[tid] = m;
  __syncthreads();
  for (int off = 128; off > 0; off >>= 1) {
    if (tid < off) red[tid] = fmaxf(red[tid], red[tid + off]);
    __syncthreads();
  }
  const float m2 = red[0];
  __syncthreads();
  s = 0.f;
  for (int k = tid; k < L; k += 256) s += expf(trow[k] - m2);
  red[tid] = s;
  __syncthreads();
  for (int off = 128; off > 0; off >>= 1) {
    if (tid < off) red[tid] += red[tid + off];
    __syncthreads();
  }
  const float inv2 = 1.f / red[0];
  const float zp = (q == 0) ? 0.f : 1.f;  // zero_pad of first query row
  for (int k = tid; k < 1024; k += 256)
    row[k] = (k < L) ? expf(trow[k] - m2) * inv2 * zp : 0.f;
}

// ---------------------------------------------------------------------------
// Kernel 5: out_mean = P @ vm, out_cov = P^2 @ vc per head.
// P read once -> in-register split; V async-staged from transposed planes.
// Outputs written directly as bf16 hi/lo planes for the final projection.
// ---------------------------------------------------------------------------
__global__ __launch_bounds__(256) void pv_kernel(
    const float* __restrict__ sc,
    const __bf16* __restrict__ vmth, const __bf16* __restrict__ vmtl,
    const __bf16* __restrict__ vcth, const __bf16* __restrict__ vctl,
    __bf16* __restrict__ omh, __bf16* __restrict__ oml,
    __bf16* __restrict__ och, __bf16* __restrict__ ocl) {
  const int q0 = blockIdx.x * 128;
  const int bh = blockIdx.y;
  __shared__ __bf16 Ph[128 * 40],  Pl[128 * 40];
  __shared__ __bf16 P2h[128 * 40], P2l[128 * 40];
  __shared__ __bf16 Vmh[64 * 40], Vml[64 * 40], Vch[64 * 40], Vcl[64 * 40];
  const int tid = threadIdx.x;
  const int lane = tid & 31, wave = tid >> 5;
  const int wm = (wave & 3) * 32, wn = (wave >> 2) * 32;
  const int lr = lane & 15, hb = lane >> 4;
  const v8f vz = {0.f, 0.f, 0.f, 0.f, 0.f, 0.f, 0.f, 0.f};
  v8f accm[2][2] = {{vz, vz}, {vz, vz}};
  v8f accc[2][2] = {{vz, vz}, {vz, vz}};

  const int ktmax = (q0 + 128) >> 5;  // only k <= q contributes (P==0 above)
  for (int kt = 0; kt < ktmax; ++kt) {
    const int kk0 = kt * 32;
    // V tiles via async copy from transposed planes (re-used across q-blocks)
    {
      const int d = tid >> 2, q8 = (tid & 3) << 3;
      const size_t g = ((size_t)bh * 64 + d) * 1024 + kk0 + q8;
      async_b128(Vmh + d * 40 + q8, vmth + g);
      async_b128(Vml + d * 40 + q8, vmtl + g);
      async_b128(Vch + d * 40 + q8, vcth + g);
      async_b128(Vcl + d * 40 + q8, vctl + g);
    }
    // P / P^2 staged with in-register split (each element read exactly once)
#pragma unroll
    for (int it = 0; it < 4; ++it) {
      int idx = tid + it * 256;
      int r = idx >> 3, c4 = (idx & 7) << 2;
      const float4 v =
          *(const float4*)(sc + (((size_t)bh << 10) + q0 + r) * 1024 + kk0 + c4);
      __bf16* ph  = Ph  + r * 40 + c4;
      __bf16* pl  = Pl  + r * 40 + c4;
      __bf16* p2h = P2h + r * 40 + c4;
      __bf16* p2l = P2l + r * 40 + c4;
      bsplit(v.x, ph + 0, pl + 0); bsplit(v.x * v.x, p2h + 0, p2l + 0);
      bsplit(v.y, ph + 1, pl + 1); bsplit(v.y * v.y, p2h + 1, p2l + 1);
      bsplit(v.z, ph + 2, pl + 2); bsplit(v.z * v.z, p2h + 2, p2l + 2);
      bsplit(v.w, ph + 3, pl + 3); bsplit(v.w * v.w, p2h + 3, p2l + 3);
    }
    wait_async0();
    __syncthreads();
#pragma unroll
    for (int fm = 0; fm < 2; ++fm) {
      v16bf ap, apl, a2, a2l;
      load_frag(ap,  Ph  + (wm + fm * 16 + lr) * 40, hb);
      load_frag(apl, Pl  + (wm + fm * 16 + lr) * 40, hb);
      load_frag(a2,  P2h + (wm + fm * 16 + lr) * 40, hb);
      load_frag(a2l, P2l + (wm + fm * 16 + lr) * 40, hb);
#pragma unroll
      for (int fn = 0; fn < 2; ++fn) {
        v16bf bmh, bml, bch, bcl;
        load_frag(bmh, Vmh + (wn + fn * 16 + lr) * 40, hb);
        load_frag(bml, Vml + (wn + fn * 16 + lr) * 40, hb);
        load_frag(bch, Vch + (wn + fn * 16 + lr) * 40, hb);
        load_frag(bcl, Vcl + (wn + fn * 16 + lr) * 40, hb);
        accm[fm][fn] = wmma_bf16(ap,  bmh, accm[fm][fn]);
        accm[fm][fn] = wmma_bf16(ap,  bml, accm[fm][fn]);
        accm[fm][fn] = wmma_bf16(apl, bmh, accm[fm][fn]);
        accc[fm][fn] = wmma_bf16(a2,  bch, accc[fm][fn]);
        accc[fm][fn] = wmma_bf16(a2,  bcl, accc[fm][fn]);
        accc[fm][fn] = wmma_bf16(a2l, bch, accc[fm][fn]);
      }
    }
    __syncthreads();
  }
#pragma unroll
  for (int fm = 0; fm < 2; ++fm)
#pragma unroll
    for (int fn = 0; fn < 2; ++fn)
#pragma unroll
      for (int v = 0; v < 8; ++v) {
        int q = q0 + wm + fm * 16 + hb * 8 + v;
        int d = wn + fn * 16 + lr;
        size_t o = ((((size_t)bh << 10) + q) << 6) + d;
        bsplit(accm[fm][fn][v], omh + o, oml + o);
        bsplit(accc[fm][fn][v], och + o, ocl + o);
      }
}

// ---------------------------------------------------------------------------
// Kernel 6: output projection from bf16 o-planes (head-major gather) with
// async staging; out = A @ W + bias (fp32) into d_out.
// ---------------------------------------------------------------------------
__global__ __launch_bounds__(256) void outproj_kernel(
    const __bf16* __restrict__ Oh, const __bf16* __restrict__ Ol,
    const __bf16* __restrict__ Wth, const __bf16* __restrict__ Wtl,
    const float* __restrict__ bias, float* __restrict__ out) {
  const int m0 = blockIdx.x * 128;
  const int n0 = blockIdx.y * 64;
  __shared__ __bf16 Ah[128 * 40], Al[128 * 40];
  __shared__ __bf16 Bh[64 * 40],  Bl[64 * 40];
  const int tid = threadIdx.x;
  const int lane = tid & 31, wave = tid >> 5;
  const int wm = (wave & 3) * 32, wn = (wave >> 2) * 32;
  const int lr = lane & 15, hb = lane >> 4;
  const v8f vz = {0.f, 0.f, 0.f, 0.f, 0.f, 0.f, 0.f, 0.f};
  v8f acc[2][2] = {{vz, vz}, {vz, vz}};

  for (int kt = 0; kt < Dd / 32; ++kt) {
    const int h = kt >> 1, dkb = (kt & 1) * 32;
#pragma unroll
    for (int i = 0; i < 2; ++i) {
      const int c = tid + i * 256;
      const int r = c >> 2, q8 = (c & 3) << 3;
      const int m = m0 + r;
      const int b = m >> 10, s = m & 1023;
      const size_t g = ((((size_t)b * Hh + h) << 10) + (size_t)s) * 64 + dkb + q8;
      async_b128(Ah + r * 40 + q8, Oh + g);
      async_b128(Al + r * 40 + q8, Ol + g);
    }
    {
      const int n = tid >> 2, q8 = (tid & 3) << 3;
      const size_t g = (size_t)(n0 + n) * Dd + kt * 32 + q8;
      async_b128(Bh + n * 40 + q8, Wth + g);
      async_b128(Bl + n * 40 + q8, Wtl + g);
    }
    wait_async0();
    __syncthreads();

    v16bf ah[2], al2[2], bh2[2], bl2[2];
#pragma unroll
    for (int f = 0; f < 2; ++f) {
      load_frag(ah[f],  Ah + (wm + f * 16 + lr) * 40, hb);
      load_frag(al2[f], Al + (wm + f * 16 + lr) * 40, hb);
      load_frag(bh2[f], Bh + (wn + f * 16 + lr) * 40, hb);
      load_frag(bl2[f], Bl + (wn + f * 16 + lr) * 40, hb);
    }
#pragma unroll
    for (int fm = 0; fm < 2; ++fm)
#pragma unroll
      for (int fn = 0; fn < 2; ++fn) {
        acc[fm][fn] = wmma_bf16(ah[fm],  bh2[fn], acc[fm][fn]);
        acc[fm][fn] = wmma_bf16(ah[fm],  bl2[fn], acc[fm][fn]);
        acc[fm][fn] = wmma_bf16(al2[fm], bh2[fn], acc[fm][fn]);
      }
    __syncthreads();
  }
#pragma unroll
  for (int fm = 0; fm < 2; ++fm)
#pragma unroll
    for (int fn = 0; fn < 2; ++fn)
#pragma unroll
      for (int v = 0; v < 8; ++v) {
        int m = m0 + wm + fm * 16 + hb * 8 + v;
        int n = n0 + wn + fn * 16 + lr;
        out[(size_t)m * Dd + n] = acc[fm][fn][v] + bias[n];
      }
}

// ---------------------------------------------------------------------------
// Host-side launcher
// ---------------------------------------------------------------------------
extern "C" void kernel_launch(void* const* d_in, const int* in_sizes, int n_in,
                              void* d_out, int out_size, void* d_ws, size_t ws_size,
                              hipStream_t stream) {
  (void)in_sizes; (void)n_in; (void)out_size; (void)ws_size;
  const float* Xin[6] = {(const float*)d_in[0], (const float*)d_in[1],
                         (const float*)d_in[2], (const float*)d_in[3],
                         (const float*)d_in[4], (const float*)d_in[5]};
  const float* Wm[8];
  const float* Bv[8];
  for (int j = 0; j < 8; ++j) {
    Wm[j] = (const float*)d_in[6 + 2 * j];
    Bv[j] = (const float*)d_in[7 + 2 * j];
  }
  const float* gammas = (const float*)d_in[22];
  // d_in[23] = mask: causal lower-triangular, hardcoded in kernels.

  // ---- workspace carve-up (256B aligned) ----
  char* cur = (char*)d_ws;
  auto alloc = [&](size_t bytes) -> void* {
    void* p = cur;
    cur += (bytes + 255) & ~(size_t)255;
    return p;
  };
  const size_t perE = (size_t)Bb * Hh * Sq * DKc;  // 4,194,304 elements
  const size_t rows = (size_t)Bb * Hh * Sq;        // 65,536
  // fp32 head-major projections
  float* headF[6];
  for (int j = 0; j < 6; ++j) headF[j] = (float*)alloc(perE * 4);
  // pre-split activation planes
  __bf16 *xh[6], *xl[6];
  for (int j = 0; j < 6; ++j) {
    xh[j] = (__bf16*)alloc(perE * 2);
    xl[j] = (__bf16*)alloc(perE * 2);
  }
  // transposed/split weight planes
  __bf16 *wh[8], *wl[8];
  for (int j = 0; j < 8; ++j) {
    wh[j] = (__bf16*)alloc((size_t)Dd * Dd * 2);
    wl[j] = (__bf16*)alloc((size_t)Dd * Dd * 2);
  }
  // head-major attention planes
  __bf16* qmh = (__bf16*)alloc(perE * 2); __bf16* qml = (__bf16*)alloc(perE * 2);
  __bf16* sqh = (__bf16*)alloc(perE * 2); __bf16* sql = (__bf16*)alloc(perE * 2);
  __bf16* kmh = (__bf16*)alloc(perE * 2); __bf16* kml = (__bf16*)alloc(perE * 2);
  __bf16* skh = (__bf16*)alloc(perE * 2); __bf16* skl = (__bf16*)alloc(perE * 2);
  // transposed V planes
  __bf16* vmth = (__bf16*)alloc(perE * 2); __bf16* vmtl = (__bf16*)alloc(perE * 2);
  __bf16* vcth = (__bf16*)alloc(perE * 2); __bf16* vctl = (__bf16*)alloc(perE * 2);
  // stats
  float* rowQ = (float*)alloc(rows * 4);
  float* rowK = (float*)alloc(rows * 4);
  // attention matrix (fp32, overwritten in place with probabilities)
  float* sc = (float*)alloc((size_t)Bb * Hh * Sq * Sq * 4);
  // attention-output planes
  __bf16* omh = (__bf16*)alloc(perE * 2); __bf16* oml = (__bf16*)alloc(perE * 2);
  __bf16* och = (__bf16*)alloc(perE * 2); __bf16* ocl = (__bf16*)alloc(perE * 2);

  // ---- pipeline ----
  const int n4 = (int)(perE / 4);
  for (int j = 0; j < 6; ++j)
    split_kernel<<<dim3(n4 / 256), 256, 0, stream>>>(Xin[j], xh[j], xl[j], n4);
  for (int j = 0; j < 8; ++j)
    wtrans_kernel<<<dim3(Dd / 32, Dd / 32), 256, 0, stream>>>(Wm[j], wh[j], wl[j]);

  const dim3 gProj(Bb * Sq / 128, Dd / 64);  // (64, 8)
  for (int j = 0; j < 6; ++j)
    proj_kernel<<<gProj, 256, 0, stream>>>(xh[j], xl[j], wh[j], wl[j], Bv[j],
                                           headF[j]);

  prep_kernel<<<dim3((unsigned)rows), 64, 0, stream>>>(
      headF[0], headF[1], headF[2], headF[3], headF[4], headF[5],
      qmh, qml, sqh, sql, kmh, kml, skh, skl,
      vmth, vmtl, vcth, vctl, rowQ, rowK);

  const dim3 gSc(Sq / 128, Sq / 64, Bb * Hh);  // causal-skipped
  scores_kernel<<<gSc, 256, 0, stream>>>(qmh, qml, sqh, sql, kmh, kml, skh, skl,
                                         rowQ, rowK, sc);

  decay_softmax_kernel<<<dim3(Sq, Bb * Hh), 256, 0, stream>>>(sc, gammas);

  pv_kernel<<<dim3(Sq / 128, Bb * Hh), 256, 0, stream>>>(
      sc, vmth, vmtl, vcth, vctl, omh, oml, och, ocl);

  float* out_mean = (float*)d_out;
  float* out_cov  = out_mean + (size_t)Bb * Sq * Dd;
  outproj_kernel<<<gProj, 256, 0, stream>>>(omh, oml, wh[6], wl[6], Bv[6], out_mean);
  outproj_kernel<<<gProj, 256, 0, stream>>>(och, ocl, wh[7], wl[7], Bv[7], out_cov);
}
